// ADDSEDiT_4097398800751
// MI455X (gfx1250) — compile-verified, matrix operand order checked by reading, not run on hardware
//
#include <hip/hip_runtime.h>

// ---------------- problem constants ----------------
#define DIM   1024
#define NL    8
#define NH    16
#define HD    64
#define BB    4
#define LL    1024
#define MROWS (BB*LL)          // 4096 token rows
#define SKIPF 0.70710678118654752440f
#define EPSF  1e-6f
#define TILEK 64

typedef __bf16 bf16_t;
typedef __attribute__((ext_vector_type(16))) __bf16 v16bf;
typedef __attribute__((ext_vector_type(8)))  float  v8f;
typedef __attribute__((ext_vector_type(4)))  unsigned int u32x4;
typedef __attribute__((ext_vector_type(4)))  int          i32x4;
typedef __attribute__((ext_vector_type(8)))  int          i32x8;

#if defined(__has_builtin)
#if __has_builtin(__builtin_amdgcn_tensor_load_to_lds)
#define HAVE_TDM 1
#endif
#endif
#ifndef HAVE_TDM
#define HAVE_TDM 0
#endif

union BFrag { v16bf v; uint4 u[2]; };

static __device__ __forceinline__ v8f wmma_bf16(const BFrag& a, const BFrag& b, v8f c) {
  // D = A(16x32 bf16) x B(32x16 bf16) + C(16x16 f32)
  return __builtin_amdgcn_wmma_f32_16x16x32_bf16(false, a.v, false, b.v, (short)0, c, false, false);
}

static __device__ __forceinline__ float gelu_tanh(float u) {
  return 0.5f * u * (1.0f + tanhf(0.7978845608028654f * (u + 0.044715f * u * u * u)));
}

// async copy of one 16-byte chunk: global -> LDS, tracked by ASYNCcnt
static __device__ __forceinline__ void async_b128(void* lds_dst, const void* gsrc) {
  unsigned lds_off = (unsigned)(unsigned long long)lds_dst;   // LDS aperture: low 32 bits = LDS byte addr
  asm volatile("global_load_async_to_lds_b128 %0, %1, off"
               :: "v"(lds_off), "v"(gsrc) : "memory");
}
// threshold waits: leave next-tile transfers in flight, retire only the oldest
static __device__ __forceinline__ void wait_async_le(int n) {
  if (n >= 2)      asm volatile("s_wait_asynccnt 0x2" ::: "memory");
  else if (n == 1) asm volatile("s_wait_asynccnt 0x1" ::: "memory");
  else             asm volatile("s_wait_asynccnt 0x0" ::: "memory");
}

// ---------------- small elementwise kernels ----------------

__global__ void k_silu(const float* __restrict__ c, const float* __restrict__ t,
                       float* __restrict__ sc) {
  int i = blockIdx.x * blockDim.x + threadIdx.x;   // BB*DIM
  if (i >= BB * DIM) return;
  float v = (c[i] + t[i]) * SKIPF;
  sc[i] = v / (1.0f + __expf(-v));
}

// mod[b][j] = silu_c[b] . adaln_w[:,j] + adaln_b[j]   (4 x 6144, tiny)
__global__ void k_adaln(const float* __restrict__ sc, const float* __restrict__ aw,
                        const float* __restrict__ ab, float* __restrict__ mod) {
  int i = blockIdx.x * blockDim.x + threadIdx.x;   // BB*6*DIM
  if (i >= BB * 6 * DIM) return;
  int b = i / (6 * DIM), j = i - b * (6 * DIM);
  const float* s = sc + b * DIM;
  float acc = ab[j];
  for (int k = 0; k < DIM; ++k) acc += s[k] * aw[(long)k * (6 * DIM) + j];
  mod[i] = acc;
}

// h = (LN(x) * (1+sc) + sh) * SKIP   -> bf16, one block per row
__global__ __launch_bounds__(256) void k_lnmod(const float* __restrict__ x,
                                               const float* __restrict__ mod,
                                               bf16_t* __restrict__ h,
                                               int shOff, int scOff) {
  int row = blockIdx.x;                 // 0..MROWS-1
  int b = row >> 10;                    // L = 1024
  const float* xr = x + (long)row * DIM;
  __shared__ float s1[256], s2[256];
  float a = 0.f, q = 0.f;
  for (int c = threadIdx.x; c < DIM; c += 256) { float v = xr[c]; a += v; q += v * v; }
  s1[threadIdx.x] = a; s2[threadIdx.x] = q;
  __syncthreads();
  for (int st = 128; st > 0; st >>= 1) {
    if ((int)threadIdx.x < st) { s1[threadIdx.x] += s1[threadIdx.x + st];
                                 s2[threadIdx.x] += s2[threadIdx.x + st]; }
    __syncthreads();
  }
  float mu  = s1[0] * (1.0f / DIM);
  float var = s2[0] * (1.0f / DIM) - mu * mu;
  float rs  = rsqrtf(var + EPSF);
  const float* md = mod + (long)b * 6 * DIM;
  for (int c = threadIdx.x; c < DIM; c += 256) {
    float v = (xr[c] - mu) * rs * (1.0f + md[scOff + c]) + md[shOff + c];
    h[(long)row * DIM + c] = (bf16_t)(v * SKIPF);
  }
}

// f32 W[K][N] -> bf16 Wt[N][K] (transposed so B-fragments are contiguous per lane)
__global__ void k_wT(const float* __restrict__ W, bf16_t* __restrict__ Wt, int K, int N) {
  long i = (long)blockIdx.x * blockDim.x + threadIdx.x;
  long tot = (long)K * N;
  if (i >= tot) return;
  int n = (int)(i / K), k = (int)(i - (long)n * K);
  Wt[i] = (bf16_t)W[(long)k * N + n];
}

// x = (x + g * o) * SKIP
__global__ void k_resid(float* __restrict__ x, const float* __restrict__ oin,
                        const float* __restrict__ mod, int gOff) {
  int idx = blockIdx.x * blockDim.x + threadIdx.x;   // MROWS*DIM
  if (idx >= MROWS * DIM) return;
  int c = idx & (DIM - 1);
  int b = idx >> 20;                                 // DIM*L = 2^20
  x[idx] = (x[idx] + mod[(long)b * 6 * DIM + gOff + c] * oin[idx]) * SKIPF;
}

// RoPE + repack: qkv f32 [M][3*DIM] -> q,k bf16 [B,H,L,64]; v -> vT bf16 [B,H,64,L]
__global__ void k_rope(const float* __restrict__ qkv, const float* __restrict__ cosb,
                       const float* __restrict__ sinb, bf16_t* __restrict__ q,
                       bf16_t* __restrict__ k, bf16_t* __restrict__ vT) {
  int idx = blockIdx.x * blockDim.x + threadIdx.x;   // B*H*L*(HD/2)
  if (idx >= BB * NH * LL * (HD / 2)) return;
  int i = idx & 31;
  int l = (idx >> 5)  & (LL - 1);
  int h = (idx >> 15) & (NH - 1);
  int b =  idx >> 19;
  long base = ((long)(b * LL + l)) * (3 * DIM) + h * HD + 2 * i;
  float q0 = qkv[base],             q1 = qkv[base + 1];
  float k0 = qkv[base + DIM],       k1 = qkv[base + DIM + 1];
  float v0 = qkv[base + 2 * DIM],   v1 = qkv[base + 2 * DIM + 1];
  float cs = cosb[l * HD + 2 * i],  sn = sinb[l * HD + 2 * i];
  long qo = ((long)((b * NH + h) * LL + l)) * HD + 2 * i;
  q[qo]     = (bf16_t)(q0 * cs - q1 * sn);
  q[qo + 1] = (bf16_t)(q1 * cs + q0 * sn);
  k[qo]     = (bf16_t)(k0 * cs - k1 * sn);
  k[qo + 1] = (bf16_t)(k1 * cs + k0 * sn);
  long vo = ((long)((b * NH + h) * HD + 2 * i)) * LL + l;
  vT[vo]      = (bf16_t)v0;
  vT[vo + LL] = (bf16_t)v1;
}

// ---------------- bf16 WMMA GEMM: C[M,N] = A[M,K] * Wt[N,K]^T + bias ----------------
// Double-buffered LDS tiles (64x64 A via ASYNCcnt loads, 128x64 B via TDM/TENSORcnt).
// Next tile is issued before computing the current one; threshold waits retire the oldest.
// EPI 0: f32 out + bias.  EPI 1: bf16 out + bias + tanh-GELU (mlp1).
template <int EPI>
__global__ __launch_bounds__(256) void k_gemm(const bf16_t* __restrict__ A,
                                              const bf16_t* __restrict__ Wt,
                                              const float* __restrict__ bias,
                                              float* __restrict__ outF,
                                              bf16_t* __restrict__ outB,
                                              int M, int N, int K) {
  __shared__ __align__(16) bf16_t At[2 * 64 * TILEK];    // 16 KB
  __shared__ __align__(16) bf16_t Bt[2 * 128 * TILEK];   // 32 KB
  const int m0 = blockIdx.y * 64, n0 = blockIdx.x * 128;
  const int tid = threadIdx.x, wave = tid >> 5, lane = tid & 31;
  const int wm = (wave >> 2) * 32, wn = (wave & 3) * 32;
  const int half = lane >> 4, ln = lane & 15, klo = half * 8;

  // stage tile kt into buffer buf (no waits here)
  auto stage = [&](int buf, int kt) {
    const int k0 = kt * TILEK;
#if HAVE_TDM
    if (wave == 0) {
      unsigned long long gaddr = (unsigned long long)(Wt + (long)n0 * K + k0);
      u32x4 g0 = { 1u,                                                       // count=1 (user D#)
                   (unsigned)(unsigned long long)(void*)(Bt + buf * 128 * TILEK),  // lds_addr
                   (unsigned)(gaddr & 0xFFFFFFFFu),
                   (unsigned)((gaddr >> 32) & 0x1FFFFFFu) | (2u << 30) };    // addr[56:32], type=2
      // group1: mask=0, data_size=1 (2B), tensor_dim0=K, tensor_dim1=N,
      //         tile_dim0=TILEK, tile_dim1=128, tensor_dim0_stride=K
      i32x8 g1 = { (int)(1u << 16),
                   (int)((unsigned)(K & 0xFFFF) << 16),
                   (int)(((unsigned)K >> 16) | ((unsigned)(N & 0xFFFF) << 16)),
                   (int)((((unsigned)N >> 16) & 0xFFFFu) | ((unsigned)TILEK << 16)),
                   (int)128,
                   (int)K,
                   0, 0 };
      i32x4 gz = { 0, 0, 0, 0 };
#if __clang_major__ >= 23
      i32x8 gz8 = { 0, 0, 0, 0, 0, 0, 0, 0 };
      __builtin_amdgcn_tensor_load_to_lds(g0, g1, gz, gz, gz8, 0);
#else
      __builtin_amdgcn_tensor_load_to_lds(g0, g1, gz, gz, 0);
#endif
    }
#else
#pragma unroll
    for (int i = 0; i < 4; ++i) {                      // 128x64 elems = 1024 chunks
      int c2 = tid + i * 256, row = c2 >> 3, ch = (c2 & 7) * 8;
      *(uint4*)(Bt + buf * 128 * TILEK + row * TILEK + ch) =
          *(const uint4*)(Wt + (long)(n0 + row) * K + k0 + ch);
    }
#endif
#pragma unroll
    for (int i = 0; i < 2; ++i) {                      // 64x64 elems = 512 chunks
      int c2 = tid + i * 256, row = c2 >> 3, ch = (c2 & 7) * 8;
      async_b128(At + buf * 64 * TILEK + row * TILEK + ch,
                 A + (long)(m0 + row) * K + k0 + ch);
    }
  };

  v8f acc[2][2];
  {
    v8f z = {};
    for (int i = 0; i < 2; ++i) for (int j = 0; j < 2; ++j) acc[i][j] = z;
  }

  const int ntiles = K / TILEK;
  stage(0, 0);
  for (int it = 0; it < ntiles; ++it) {
    const int cur = it & 1;
    const bool more = (it + 1) < ntiles;
    if (more) stage(cur ^ 1, it + 1);      // overlap next tile's DMA with this tile's compute
    wait_async_le(more ? 2 : 0);           // oldest A-tile loads retired; next still in flight
#if HAVE_TDM
    if (wave == 0) {
      if (more) __builtin_amdgcn_s_wait_tensorcnt(1);
      else      __builtin_amdgcn_s_wait_tensorcnt(0);
    }
#endif
    __syncthreads();

    const bf16_t* Ab = At + cur * 64 * TILEK;
    const bf16_t* Bb = Bt + cur * 128 * TILEK;
#pragma unroll
    for (int kk = 0; kk < TILEK; kk += 32) {
      BFrag a[2], b[2];
#pragma unroll
      for (int i = 0; i < 2; ++i) {
        const bf16_t* p = Ab + (wm + i * 16 + ln) * TILEK + kk;
        a[i].u[0] = *(const uint4*)(p + klo);
        a[i].u[1] = *(const uint4*)(p + klo + 16);
      }
#pragma unroll
      for (int j = 0; j < 2; ++j) {
        const bf16_t* p = Bb + (wn + j * 16 + ln) * TILEK + kk;
        b[j].u[0] = *(const uint4*)(p + klo);
        b[j].u[1] = *(const uint4*)(p + klo + 16);
      }
#pragma unroll
      for (int i = 0; i < 2; ++i)
#pragma unroll
        for (int j = 0; j < 2; ++j)
          acc[i][j] = wmma_bf16(a[i], b[j], acc[i][j]);
    }
    __syncthreads();                       // WAR: buffer reused next-next iteration
  }

  // epilogue: C layout -> VGPR r: lanes0-15 M=r, lanes16-31 M=r+8; N = lane&15
#pragma unroll
  for (int i = 0; i < 2; ++i)
#pragma unroll
    for (int j = 0; j < 2; ++j)
#pragma unroll
      for (int r = 0; r < 8; ++r) {
        int row = m0 + wm + i * 16 + r + half * 8;
        int col = n0 + wn + j * 16 + ln;
        float v = acc[i][j][r] + bias[col];
        if (EPI == 1) outB[(long)row * N + col] = (bf16_t)gelu_tanh(v);
        else          outF[(long)row * N + col] = v;
      }
}

// ---------------- attention: one wave per (b,h,16-query tile) ----------------
__global__ __launch_bounds__(32) void k_attn(const bf16_t* __restrict__ q,
                                             const bf16_t* __restrict__ kbuf,
                                             const bf16_t* __restrict__ vT,
                                             bf16_t* __restrict__ o) {
  extern __shared__ char smem[];
  float*  s = (float*)smem;                       // 16 x 1024 f32 scores (64 KB)
  bf16_t* p = (bf16_t*)(smem + 16 * LL * 4);      // 16 x 1024 bf16 probs (32 KB)

  int blk = blockIdx.x;
  int qt = blk & (LL / 16 - 1);
  int bh = blk >> 6;                              // LL/16 == 64
  int b = bh >> 4, h = bh & 15;
  int lane = threadIdx.x, half = lane >> 4, ln = lane & 15, klo = half * 8;
  int q0r = qt * 16;

  // Q fragments (16 rows x 64 dims -> two 16x32 A frags), contiguous per lane
  const bf16_t* qrow = q + ((long)(bh * LL + q0r + ln)) * HD;
  BFrag a0, a1;
  a0.u[0] = *(const uint4*)(qrow + klo);       a0.u[1] = *(const uint4*)(qrow + klo + 16);
  a1.u[0] = *(const uint4*)(qrow + 32 + klo);  a1.u[1] = *(const uint4*)(qrow + 32 + klo + 16);

  const float scale = 0.125f;                   // 64^-0.5
  for (int kt = 0; kt < LL / 16; ++kt) {
    const bf16_t* krow = kbuf + ((long)(bh * LL + kt * 16 + ln)) * HD;
    BFrag b0, b1;
    b0.u[0] = *(const uint4*)(krow + klo);       b0.u[1] = *(const uint4*)(krow + klo + 16);
    b1.u[0] = *(const uint4*)(krow + 32 + klo);  b1.u[1] = *(const uint4*)(krow + 32 + klo + 16);
    v8f acc = {};
    acc = wmma_bf16(a0, b0, acc);
    acc = wmma_bf16(a1, b1, acc);
#pragma unroll
    for (int r = 0; r < 8; ++r)
      s[(r + half * 8) * LL + kt * 16 + ln] = acc[r] * scale;
  }
  __syncthreads();

  // exact softmax per row (wave32 shuffle reductions)
  for (int r = 0; r < 16; ++r) {
    float m = -1e30f;
    for (int c = lane; c < LL; c += 32) m = fmaxf(m, s[r * LL + c]);
#pragma unroll
    for (int off = 16; off > 0; off >>= 1) m = fmaxf(m, __shfl_xor(m, off, 32));
    float sum = 0.f;
    for (int c = lane; c < LL; c += 32) { float e = __expf(s[r * LL + c] - m); s[r * LL + c] = e; sum += e; }
#pragma unroll
    for (int off = 16; off > 0; off >>= 1) sum += __shfl_xor(sum, off, 32);
    float inv = 1.0f / sum;
    for (int c = lane; c < LL; c += 32) p[r * LL + c] = (bf16_t)(s[r * LL + c] * inv);
  }
  __syncthreads();

  // O(16x64) = P(16x1024) x V(1024x64), vT is [B,H,64,L] so B frags are contiguous
  v8f oa[4];
  { v8f z = {}; for (int j = 0; j < 4; ++j) oa[j] = z; }
  for (int kb = 0; kb < LL; kb += 32) {
    BFrag pa;
    pa.u[0] = *(const uint4*)(p + ln * LL + kb + klo);
    pa.u[1] = *(const uint4*)(p + ln * LL + kb + klo + 16);
#pragma unroll
    for (int j = 0; j < 4; ++j) {
      const bf16_t* vrow = vT + ((long)(bh * HD + j * 16 + ln)) * LL + kb;
      BFrag vb;
      vb.u[0] = *(const uint4*)(vrow + klo);
      vb.u[1] = *(const uint4*)(vrow + klo + 16);
      oa[j] = wmma_bf16(pa, vb, oa[j]);
    }
  }
#pragma unroll
  for (int j = 0; j < 4; ++j)
#pragma unroll
    for (int r = 0; r < 8; ++r) {
      int row = q0r + r + half * 8;
      int col = h * HD + j * 16 + ln;
      o[((long)(b * LL + row)) * DIM + col] = (bf16_t)oa[j][r];
    }
}

// ---------------- host orchestration ----------------
extern "C" void kernel_launch(void* const* d_in, const int* in_sizes, int n_in,
                              void* d_out, int out_size, void* d_ws, size_t ws_size,
                              hipStream_t stream) {
  const float* x_in   = (const float*)d_in[0];
  const float* c_in   = (const float*)d_in[1];
  const float* t_in   = (const float*)d_in[2];
  const float* cosb   = (const float*)d_in[4];
  const float* sinb   = (const float*)d_in[5];
  const float* qkv_w  = (const float*)d_in[6];
  const float* qkv_b  = (const float*)d_in[7];
  const float* proj_w = (const float*)d_in[8];
  const float* proj_b = (const float*)d_in[9];
  const float* w1     = (const float*)d_in[10];
  const float* b1     = (const float*)d_in[11];
  const float* w2     = (const float*)d_in[12];
  const float* b2     = (const float*)d_in[13];
  const float* aw     = (const float*)d_in[14];
  const float* ab     = (const float*)d_in[15];

  char* ws = (char*)d_ws;
  size_t off = 0;
  auto alloc = [&](size_t bytes) -> char* {
    char* pp = ws + off;
    off = (off + bytes + 255) & ~(size_t)255;
    return pp;
  };
  float*  X   = (float*) alloc((size_t)MROWS * DIM * 4);       // running activations
  float*  SC  = (float*) alloc((size_t)BB * DIM * 4);          // silu(cmod)
  float*  MD  = (float*) alloc((size_t)BB * 6 * DIM * 4);      // adaLN mod (per layer)
  bf16_t* Hb  = (bf16_t*)alloc((size_t)MROWS * DIM * 2);       // LN output
  float*  QKV = (float*) alloc((size_t)MROWS * 3 * DIM * 4);   // qkv out / reused as G
  bf16_t* Qb  = (bf16_t*)alloc((size_t)MROWS * DIM * 2);
  bf16_t* Kb  = (bf16_t*)alloc((size_t)MROWS * DIM * 2);
  bf16_t* VT  = (bf16_t*)alloc((size_t)MROWS * DIM * 2);
  bf16_t* Ob  = (bf16_t*)alloc((size_t)MROWS * DIM * 2);
  float*  P1  = (float*) alloc((size_t)MROWS * DIM * 4);       // proj / mlp2 out
  bf16_t* WT  = (bf16_t*)alloc((size_t)4 * DIM * DIM * 2);     // converted weights
  bf16_t* G   = (bf16_t*)QKV;                                  // gelu out reuses QKV space

  hipMemcpyAsync(X, x_in, (size_t)MROWS * DIM * 4, hipMemcpyDeviceToDevice, stream);
  k_silu<<<(BB * DIM + 255) / 256, 256, 0, stream>>>(c_in, t_in, SC);

  const size_t attn_lds = (size_t)16 * LL * 4 + (size_t)16 * LL * 2;  // 96 KB

  for (int l = 0; l < NL; ++l) {
    k_adaln<<<(BB * 6 * DIM + 255) / 256, 256, 0, stream>>>(
        SC, aw + (size_t)l * DIM * 6 * DIM, ab + (size_t)l * 6 * DIM, MD);

    k_lnmod<<<MROWS, 256, 0, stream>>>(X, MD, Hb, 0, DIM);

    { long tot = (long)DIM * 3 * DIM;
      k_wT<<<(int)((tot + 255) / 256), 256, 0, stream>>>(qkv_w + (size_t)l * DIM * 3 * DIM, WT, DIM, 3 * DIM); }
    k_gemm<0><<<dim3(3 * DIM / 128, MROWS / 64), 256, 0, stream>>>(
        Hb, WT, qkv_b + (size_t)l * 3 * DIM, QKV, nullptr, MROWS, 3 * DIM, DIM);

    k_rope<<<(BB * NH * LL * (HD / 2) + 255) / 256, 256, 0, stream>>>(QKV, cosb, sinb, Qb, Kb, VT);

    k_attn<<<BB * NH * (LL / 16), 32, attn_lds, stream>>>(Qb, Kb, VT, Ob);

    { long tot = (long)DIM * DIM;
      k_wT<<<(int)((tot + 255) / 256), 256, 0, stream>>>(proj_w + (size_t)l * DIM * DIM, WT, DIM, DIM); }
    k_gemm<0><<<dim3(DIM / 128, MROWS / 64), 256, 0, stream>>>(
        Ob, WT, proj_b + (size_t)l * DIM, P1, nullptr, MROWS, DIM, DIM);

    k_resid<<<(MROWS * DIM + 255) / 256, 256, 0, stream>>>(X, P1, MD, 2 * DIM);

    k_lnmod<<<MROWS, 256, 0, stream>>>(X, MD, Hb, 3 * DIM, 4 * DIM);

    { long tot = (long)DIM * 4 * DIM;
      k_wT<<<(int)((tot + 255) / 256), 256, 0, stream>>>(w1 + (size_t)l * DIM * 4 * DIM, WT, DIM, 4 * DIM); }
    k_gemm<1><<<dim3(4 * DIM / 128, MROWS / 64), 256, 0, stream>>>(
        Hb, WT, b1 + (size_t)l * 4 * DIM, nullptr, G, MROWS, 4 * DIM, DIM);

    { long tot = (long)4 * DIM * DIM;
      k_wT<<<(int)((tot + 255) / 256), 256, 0, stream>>>(w2 + (size_t)l * 4 * DIM * DIM, WT, 4 * DIM, DIM); }
    k_gemm<0><<<dim3(DIM / 128, MROWS / 64), 256, 0, stream>>>(
        G, WT, b2 + (size_t)l * DIM, P1, nullptr, MROWS, DIM, 4 * DIM);

    k_resid<<<(MROWS * DIM + 255) / 256, 256, 0, stream>>>(X, P1, MD, 5 * DIM);
  }

  hipMemcpyAsync(d_out, X, (size_t)MROWS * DIM * 4, hipMemcpyDeviceToDevice, stream);
}
